// SPA_4982162063813
// MI455X (gfx1250) — matrix operand order repaired and checked by reading, not compiled, not run on hardware
//
#include <hip/hip_runtime.h>
#include <cstdint>

// ---------------- problem constants (from reference) ----------------
#define B_    2
#define C_    64
#define H_    128
#define W_    128
#define HW_   (H_ * W_)          // 16384
#define QKD   64
#define HEADS 2
#define HD    32                 // head dim (== WMMA K for f16!)
#define KSP   256
#define T_    256
#define SCALE_F 0.17677669529663689f   // 32^-0.5
#define LN_EPS  1e-6f

// LDS pitches (halves) chosen to break bank conflicts
#define PQ  40     // Q/K row pitch  (80 B  -> 16B-chunk reads hit distinct bank groups)
#define PVT 264    // V^T row pitch  (528 B)
#define PP  264    // P   row pitch  (528 B)

typedef __attribute__((ext_vector_type(16))) _Float16 v16h;
typedef __attribute__((ext_vector_type(8)))  _Float16 v8h;
typedef __attribute__((ext_vector_type(8)))  float    v8f;

__device__ __forceinline__ v16h cat16(v8h lo, v8h hi) {
  return __builtin_shufflevector(lo, hi, 0,1,2,3,4,5,6,7,8,9,10,11,12,13,14,15);
}

// =====================================================================
// Kernel 1: per-pixel channel LayerNorm + Q/K/V projections.
// Block = 256 threads handles 64 consecutive pixels.
// Outputs stored pixel-major: qT/kT [b][n][o], vT [b][n][c] so the
// attention gather reads 128 contiguous bytes per row.
// =====================================================================
__global__ void spa_ln_qkv(const float* __restrict__ x,
                           const float* __restrict__ ln_g,
                           const float* __restrict__ ln_b,
                           const float* __restrict__ Wq,
                           const float* __restrict__ Wk,
                           const float* __restrict__ Wv,
                           float* __restrict__ qT,
                           float* __restrict__ kT,
                           float* __restrict__ vT) {
  __shared__ float xs[64][65];     // padded: bank-conflict-free column reads
  __shared__ float smu[64], srstd[64];

  const int blk = blockIdx.x;                 // B*HW/64 = 512 blocks
  const int b   = blk / (HW_ / 64);
  const int n0  = (blk % (HW_ / 64)) * 64;
  const int t   = threadIdx.x;

  // coalesced load: consecutive threads -> consecutive pixels of one channel
  for (int i = t; i < 64 * 64; i += 256) {
    const int c = i >> 6, p = i & 63;
    xs[c][p] = x[((size_t)b * C_ + c) * HW_ + n0 + p];
  }
  __syncthreads();

  if (t < 64) {                               // one thread per pixel: LN stats
    float s = 0.f, s2 = 0.f;
    for (int c = 0; c < 64; ++c) { const float v = xs[c][t]; s += v; s2 += v * v; }
    const float mu  = s * (1.0f / 64.0f);
    const float var = s2 * (1.0f / 64.0f) - mu * mu;
    smu[t] = mu;  srstd[t] = rsqrtf(var + LN_EPS);
  }
  __syncthreads();

  for (int i = t; i < 64 * 64; i += 256) {
    const int c = i >> 6, p = i & 63;
    xs[c][p] = (xs[c][p] - smu[p]) * srstd[p] * ln_g[c] + ln_b[c];
  }
  __syncthreads();

  // projections: thread -> (output channel o, pixel group of 16)
  const int o  = t & 63;
  const int pg = t >> 6;
  float accq[16], acck[16], accv[16];
  #pragma unroll
  for (int j = 0; j < 16; ++j) { accq[j] = acck[j] = accv[j] = 0.f; }

  for (int c = 0; c < 64; ++c) {
    const float wq = Wq[o * 64 + c];
    const float wk = Wk[o * 64 + c];
    const float wv = Wv[o * 64 + c];
    #pragma unroll
    for (int j = 0; j < 16; ++j) {
      const float xv = xs[c][pg * 16 + j];    // same address for whole wave: LDS broadcast
      accq[j] = fmaf(wq, xv, accq[j]);
      acck[j] = fmaf(wk, xv, acck[j]);
      accv[j] = fmaf(wv, xv, accv[j]);
    }
  }
  #pragma unroll
  for (int j = 0; j < 16; ++j) {
    const size_t n = (size_t)b * HW_ + n0 + pg * 16 + j;
    qT[n * QKD + o] = accq[j];
    kT[n * QKD + o] = acck[j];
    vT[n * C_  + o] = accv[j];
  }
}

// =====================================================================
// Kernel 2: exact f32 per-pixel-per-head ||q||^2, ||k||^2
// =====================================================================
__global__ void spa_norms(const float* __restrict__ qT, const float* __restrict__ kT,
                          float* __restrict__ q2, float* __restrict__ k2) {
  const int i = blockIdx.x * blockDim.x + threadIdx.x;   // B*HEADS*HW
  if (i >= B_ * HEADS * HW_) return;
  const int n  = i % HW_;
  const int h  = (i / HW_) % HEADS;
  const int b  = i / (HW_ * HEADS);
  const float* qr = qT + ((size_t)b * HW_ + n) * QKD + h * HD;
  const float* kr = kT + ((size_t)b * HW_ + n) * QKD + h * HD;
  float sq = 0.f, sk = 0.f;
  #pragma unroll
  for (int j = 0; j < HD; ++j) { sq = fmaf(qr[j], qr[j], sq); sk = fmaf(kr[j], kr[j], sk); }
  q2[i] = sq; k2[i] = sk;
}

// =====================================================================
// Kernel 3: segment counts (how many of the K_SP*T gathered slots map to
// each pixel)
// =====================================================================
__global__ void spa_count(const int* __restrict__ indices, float* __restrict__ cnt) {
  const int i = blockIdx.x * blockDim.x + threadIdx.x;   // B*KSP*T
  if (i >= B_ * KSP * T_) return;
  const int b = i / (KSP * T_);
  atomicAdd(&cnt[(size_t)b * HW_ + indices[i]], 1.0f);
}

// =====================================================================
// Kernel 4: distance attention per (b, k_sp, head). 1024 blocks, 256 thr.
//   S = Q K^T via v_wmma_f32_16x16x32_f16 (one WMMA per tile: HD==32==K)
//   d2 = q2 + k2 - 2 S  (q2/k2 exact f32) -> logits -> softmax in regs
//   O = P (wgt*V) via WMMA, 8 accumulating WMMAs per 16x16 output tile
//   scatter wgt*O with f32 global atomics.
// =====================================================================
__global__ void __launch_bounds__(256, 1)
spa_attn(const float* __restrict__ qT, const float* __restrict__ kT,
         const float* __restrict__ vT,
         const float* __restrict__ q2g, const float* __restrict__ k2g,
         const float* __restrict__ sims, const int* __restrict__ indices,
         const unsigned char* __restrict__ mask,      // jnp bool -> 1 byte
         float* __restrict__ sums) {
  extern __shared__ float4 smem4[];
  _Float16* sQ   = (_Float16*)smem4;            // 256 * PQ
  _Float16* sK   = sQ  + 256 * PQ;              // 256 * PQ
  _Float16* sVT  = sK  + 256 * PQ;              // 32  * PVT   (wgt-scaled V, transposed)
  _Float16* sP   = sVT + 32  * PVT;             // 8 waves * 16 * PP
  float*    sSim = (float*)(sP + 8 * 16 * PP);  // 256
  float*    sQ2  = sSim + 256;                  // 256
  float*    sK2  = sQ2  + 256;                  // 256
  float*    sMB  = sK2  + 256;                  // 256 additive mask bias
  int*      sIdx = (int*)(sMB + 256);           // 256

  const int blk = blockIdx.x;                   // b*KSP*HEADS + kk*HEADS + h
  const int h   = blk % HEADS;
  const int kk  = (blk / HEADS) % KSP;
  const int b   = blk / (HEADS * KSP);
  const int t   = threadIdx.x;

  // ---- gather phase: row s = t ----
  {
    const size_t base = ((size_t)b * KSP + kk) * T_ + t;
    const int pix = indices[base];
    sIdx[t] = pix;
    const float w = sims[base];
    sSim[t] = w;
    sMB[t]  = mask[base] ? 0.0f : -1e30f;
    sQ2[t]  = q2g[((size_t)b * HEADS + h) * HW_ + pix];
    sK2[t]  = k2g[((size_t)b * HEADS + h) * HW_ + pix];
    const float* qr = qT + ((size_t)b * HW_ + pix) * QKD + h * HD;
    const float* kr = kT + ((size_t)b * HW_ + pix) * QKD + h * HD;
    const float* vr = vT + ((size_t)b * HW_ + pix) * C_  + h * HD;
    #pragma unroll
    for (int j = 0; j < HD; ++j) {
      sQ[t * PQ + j]  = (_Float16)qr[j];
      sK[t * PQ + j]  = (_Float16)kr[j];
      sVT[j * PVT + t] = (_Float16)(w * vr[j]);   // inner wgt*V, transposed
    }
  }
  __syncthreads();

  const int wave = t >> 5;
  const int lane = t & 31;
  const bool hi  = lane >= 16;
  const int l16  = lane & 15;
  _Float16* sPw  = sP + wave * 16 * PP;

  // each wave owns row tiles {wave, wave+8}  (16 tiles cover T=256 rows)
  for (int ii = 0; ii < 2; ++ii) {
    const int it = wave + ii * 8;
    const int rb = it * 16;

    // A = Q tile (16x32 f16) k = 16*half + s (contiguous, same as B)
    const _Float16* qr = sQ + (rb + l16) * PQ;
    const v16h a = cat16(*(const v8h*)(qr + (hi ? 16 : 0)),
                         *(const v8h*)(qr + (hi ? 24 : 8)));

    // ---- pass 1: full 16x256 score strip in registers ----
    v8f S[16];
    #pragma unroll
    for (int jt = 0; jt < 16; ++jt) {
      const _Float16* krp = sK + (jt * 16 + l16) * PQ + (hi ? 16 : 0);
      const v16h bK = cat16(*(const v8h*)krp, *(const v8h*)(krp + 8));
      const v8f cz = {};
      S[jt] = __builtin_amdgcn_wmma_f32_16x16x32_f16(false, a, false, bK,
                                                     (short)0, cz, false, false);
    }

    // ---- d2 -> logits, track row max ----
    float rmax[8];
    #pragma unroll
    for (int vv = 0; vv < 8; ++vv) rmax[vv] = -3.0e38f;
    #pragma unroll
    for (int jt = 0; jt < 16; ++jt) {
      #pragma unroll
      for (int vv = 0; vv < 8; ++vv) {
        const int scol = jt * 16 + l16;
        const float qq = sQ2[rb + vv + (hi ? 8 : 0)];   // broadcast within half-wave
        float d2 = qq + sK2[scol] - 2.0f * S[jt][vv];
        d2 = fmaxf(d2, 0.0f);
        const float lg = -SCALE_F * sqrtf(d2) + sMB[scol];
        S[jt][vv] = lg;
        rmax[vv] = fmaxf(rmax[vv], lg);
      }
    }
    #pragma unroll
    for (int vv = 0; vv < 8; ++vv) {    // reduce across the 16 lanes of each half
      float m = rmax[vv];
      m = fmaxf(m, __shfl_xor(m, 1)); m = fmaxf(m, __shfl_xor(m, 2));
      m = fmaxf(m, __shfl_xor(m, 4)); m = fmaxf(m, __shfl_xor(m, 8));
      rmax[vv] = m;
    }

    // ---- exp + row sum ----
    float rsum[8];
    #pragma unroll
    for (int vv = 0; vv < 8; ++vv) rsum[vv] = 0.f;
    #pragma unroll
    for (int jt = 0; jt < 16; ++jt) {
      #pragma unroll
      for (int vv = 0; vv < 8; ++vv) {
        const float e = __expf(S[jt][vv] - rmax[vv]);
        S[jt][vv] = e;
        rsum[vv] += e;
      }
    }
    #pragma unroll
    for (int vv = 0; vv < 8; ++vv) {
      float s = rsum[vv];
      s += __shfl_xor(s, 1); s += __shfl_xor(s, 2);
      s += __shfl_xor(s, 4); s += __shfl_xor(s, 8);
      rsum[vv] = 1.0f / s;              // all-masked row -> uniform, matches jax
    }

    // ---- store normalized P (f16) to this wave's LDS strip ----
    #pragma unroll
    for (int jt = 0; jt < 16; ++jt) {
      #pragma unroll
      for (int vv = 0; vv < 8; ++vv) {
        const int r = vv + (hi ? 8 : 0);
        sPw[r * PP + jt * 16 + l16] = (_Float16)(S[jt][vv] * rsum[vv]);
      }
    }
    __syncthreads();   // uniform across block (all waves run both ii iters)

    // ---- O = P * (wgt*V): accumulate 8 WMMAs per 16-wide column tile ----
    v8f o0 = {}, o1 = {};
    #pragma unroll
    for (int st = 0; st < 8; ++st) {
      const _Float16* pr = sPw + l16 * PP + st * 32;
      const v16h ap = cat16(*(const v8h*)(pr + (hi ? 16 : 0)),
                            *(const v8h*)(pr + (hi ? 24 : 8)));
      const _Float16* v0p = sVT + l16 * PVT        + st * 32 + (hi ? 16 : 0);
      const _Float16* v1p = sVT + (16 + l16) * PVT + st * 32 + (hi ? 16 : 0);
      const v16h b0 = cat16(*(const v8h*)v0p, *(const v8h*)(v0p + 8));
      const v16h b1 = cat16(*(const v8h*)v1p, *(const v8h*)(v1p + 8));
      o0 = __builtin_amdgcn_wmma_f32_16x16x32_f16(false, ap, false, b0, (short)0, o0, false, false);
      o1 = __builtin_amdgcn_wmma_f32_16x16x32_f16(false, ap, false, b1, (short)0, o1, false, false);
    }

    // ---- outer wgt multiply + scatter-add ----
    #pragma unroll
    for (int vv = 0; vv < 8; ++vv) {
      const int r   = rb + vv + (hi ? 8 : 0);
      const float w = sSim[r];
      const int pix = sIdx[r];
      const int c0  = h * HD + l16;
      atomicAdd(&sums[((size_t)b * C_ + c0)      * HW_ + pix], o0[vv] * w);
      atomicAdd(&sums[((size_t)b * C_ + c0 + 16) * HW_ + pix], o1[vv] * w);
    }
    __syncthreads();   // protect sPw before next ii overwrites it
  }
}

// =====================================================================
// Kernel 5: segment mean + fallback-to-v blend, write (B,C,H,W) output
// =====================================================================
__global__ void spa_finalize(const float* __restrict__ sums, const float* __restrict__ cnt,
                             const float* __restrict__ vT, float* __restrict__ out) {
  const int i = blockIdx.x * blockDim.x + threadIdx.x;   // B*C*HW
  if (i >= B_ * C_ * HW_) return;
  const int n = i % HW_;
  const int c = (i / HW_) % C_;
  const int b = i / (HW_ * C_);
  const float cv = cnt[(size_t)b * HW_ + n];
  out[i] = (cv > 0.f) ? (sums[i] / cv)
                      : vT[((size_t)b * HW_ + n) * C_ + c];
}

// =====================================================================
// host-side launch
// =====================================================================
extern "C" void kernel_launch(void* const* d_in, const int* in_sizes, int n_in,
                              void* d_out, int out_size, void* d_ws, size_t ws_size,
                              hipStream_t stream) {
  (void)in_sizes; (void)n_in; (void)out_size; (void)ws_size;

  const float*         x    = (const float*)d_in[0];
  const float*         sims = (const float*)d_in[1];
  const int*           idx  = (const int*)d_in[2];
  const unsigned char* mask = (const unsigned char*)d_in[3];  // jnp bool = 1B
  const float*         ln_g = (const float*)d_in[4];
  const float*         ln_b = (const float*)d_in[5];
  const float*         Wq   = (const float*)d_in[6];
  const float*         Wk   = (const float*)d_in[7];
  const float*         Wv   = (const float*)d_in[8];
  float* out = (float*)d_out;

  // workspace carve-up (~33 MB total, all f32)
  float* ws  = (float*)d_ws;
  float* qT  = ws;                                    // B*HW*QKD
  float* kT  = qT  + (size_t)B_ * HW_ * QKD;          // B*HW*QKD
  float* vT  = kT  + (size_t)B_ * HW_ * QKD;          // B*HW*C
  float* q2  = vT  + (size_t)B_ * HW_ * C_;           // B*HEADS*HW
  float* k2  = q2  + (size_t)B_ * HEADS * HW_;        // B*HEADS*HW
  float* cnt = k2  + (size_t)B_ * HEADS * HW_;        // B*HW
  float* sums = cnt + (size_t)B_ * HW_;               // B*C*HW (contiguous after cnt)

  // zero accumulators every call (graph-capture-safe memset node)
  hipMemsetAsync(cnt, 0, (size_t)(B_ * HW_ + B_ * C_ * HW_) * sizeof(float), stream);

  spa_ln_qkv<<<B_ * HW_ / 64, 256, 0, stream>>>(x, ln_g, ln_b, Wq, Wk, Wv, qT, kT, vT);
  spa_norms<<<(B_ * HEADS * HW_ + 255) / 256, 256, 0, stream>>>(qT, kT, q2, k2);
  spa_count<<<(B_ * KSP * T_ + 255) / 256, 256, 0, stream>>>(idx, cnt);

  constexpr size_t SMEM_BYTES =
      (size_t)(256 * PQ + 256 * PQ + 32 * PVT + 8 * 16 * PP) * sizeof(_Float16)
      + 4 * 256 * sizeof(float) + 256 * sizeof(int);   // = 130,560 B
  spa_attn<<<B_ * KSP * HEADS, 256, SMEM_BYTES, stream>>>(qT, kT, vT, q2, k2,
                                                          sims, idx, mask, sums);

  spa_finalize<<<(B_ * C_ * HW_ + 255) / 256, 256, 0, stream>>>(sums, cnt, vT, out);
}
